// DSTP_rnn_75170517614885
// MI455X (gfx1250) — matrix operand order repaired and checked by reading, not compiled, not run on hardware
//
#include <hip/hip_runtime.h>
#include <cstdint>
#include <cstddef>

typedef __attribute__((ext_vector_type(16))) __bf16 v16bf;
typedef __attribute__((ext_vector_type(8)))  float  v8f;
typedef __attribute__((ext_vector_type(4)))  unsigned int v4u;

#define HE     128
#define NIN    81
#define NINP   96
#define TSTEPS 32
#define BROWS  16
#define BSZ    1024

__device__ __forceinline__ float sigf(float x) { return 1.0f / (1.0f + __expf(-x)); }

__device__ __forceinline__ v8f wmma_bf16(v16bf a, v16bf b, v8f c) {
  // D = A(16x32) * B(32x16) + C, f32 accumulate
  return __builtin_amdgcn_wmma_f32_16x16x32_bf16(false, a, false, b, (short)0, c, false, false);
}

// ---------------------------------------------------------------------------
// B-fragment streaming via inline asm: two global_load_b128 per 32x16 bf16
// fragment. asm volatile pins the loads inside the recurrent t-loop (no LICM
// hoist -> no scratch spills). Early-clobber outputs ("=&v") keep the dest
// registers disjoint from the address register. The wait carries a
// read-write dependency on every fragment so consumers can NOT be scheduled
// before the wait and fragments can NOT be register-coalesced across it.
// ---------------------------------------------------------------------------
struct bfrag { v4u lo, hi; };

__device__ __forceinline__ void issue_b(const __bf16* p, bfrag& f) {
  asm volatile("global_load_b128 %0, %2, off\n\t"
               "global_load_b128 %1, %2, off offset:16"
               : "=&v"(f.lo), "=&v"(f.hi)
               : "v"(p)
               : "memory");
}

__device__ __forceinline__ void wait_frags4(bfrag& a, bfrag& b, bfrag& c, bfrag& d) {
  asm volatile("s_wait_loadcnt 0x0"
               : "+v"(a.lo), "+v"(a.hi), "+v"(b.lo), "+v"(b.hi),
                 "+v"(c.lo), "+v"(c.hi), "+v"(d.lo), "+v"(d.hi)
               :: "memory");
}
__device__ __forceinline__ void wait_frags7(bfrag& a, bfrag& b, bfrag& c, bfrag& d,
                                            bfrag& e, bfrag& f, bfrag& g) {
  asm volatile("s_wait_loadcnt 0x0"
               : "+v"(a.lo), "+v"(a.hi), "+v"(b.lo), "+v"(b.hi),
                 "+v"(c.lo), "+v"(c.hi), "+v"(d.lo), "+v"(d.hi),
                 "+v"(e.lo), "+v"(e.hi), "+v"(f.lo), "+v"(f.hi),
                 "+v"(g.lo), "+v"(g.hi)
               :: "memory");
}
__device__ __forceinline__ void wait_frags8(bfrag& a, bfrag& b, bfrag& c, bfrag& d,
                                            bfrag& e, bfrag& f, bfrag& g, bfrag& h) {
  asm volatile("s_wait_loadcnt 0x0"
               : "+v"(a.lo), "+v"(a.hi), "+v"(b.lo), "+v"(b.hi),
                 "+v"(c.lo), "+v"(c.hi), "+v"(d.lo), "+v"(d.hi),
                 "+v"(e.lo), "+v"(e.hi), "+v"(f.lo), "+v"(f.hi),
                 "+v"(g.lo), "+v"(g.hi), "+v"(h.lo), "+v"(h.hi)
               :: "memory");
}

__device__ __forceinline__ v16bf as_v16bf(const bfrag& f) {
  return __builtin_bit_cast(v16bf, f);
}

// A-fragment (16x32 bf16) from row-major [m][k] tile, leading dim ldk (LDS or global).
// Layout: lanes 0-15 -> M=lane, lanes 16-31 -> M=lane-16;
// elems 0..15 -> K = (lane>=16?16:0)+e (contiguous, matches B fragments)
__device__ __forceinline__ v16bf load_a(const __bf16* tile, int ldk, int lane) {
  int m  = lane & 15;
  int hs = (lane >> 4) & 1;
  const __bf16* p = tile + m * ldk + hs * 16;
  v16bf a;
#pragma unroll
  for (int e = 0; e < 8; ++e) a[e] = p[e];
#pragma unroll
  for (int e = 0; e < 8; ++e) a[8 + e] = p[8 + e];
  return a;
}

// Plain-C++ B-fragment load (used only in the one-shot epre kernel).
__device__ __forceinline__ v16bf load_b(const __bf16* tile, int ldk, int lane) {
  int n  = lane & 15;
  int hs = (lane >> 4) & 1;
  const __bf16* p = tile + n * ldk + hs * 16;
  v16bf b;
#pragma unroll
  for (int e = 0; e < 16; ++e) b[e] = p[e];
  return b;
}

// ---------------------------------------------------------------------------
// Kernel 1: convert / pad weights to bf16 in workspace
// ---------------------------------------------------------------------------
__global__ __launch_bounds__(256) void dstp_prep(
    const float* __restrict__ l1Wih, const float* __restrict__ l1Whh,
    const float* __restrict__ l2Wih, const float* __restrict__ l2Whh,
    const float* __restrict__ dWhh,  const float* __restrict__ decW1,
    __bf16* __restrict__ W1ih, __bf16* __restrict__ W1hh,
    __bf16* __restrict__ W2ih, __bf16* __restrict__ W2hh,
    __bf16* __restrict__ DWhh,
    __bf16* __restrict__ W1e, __bf16* __restrict__ W1d, __bf16* __restrict__ W1c)
{
  const int total = 49152 * 2 + 65536 * 3 + 16384 * 3;
  for (int i = blockIdx.x * 256 + threadIdx.x; i < total; i += gridDim.x * 256) {
    int idx = i;
    if (idx < 49152) { int n = idx / 96, k = idx - n * 96;
      W1ih[idx] = (__bf16)(k < NIN ? l1Wih[n * NIN + k] : 0.f); continue; }
    idx -= 49152;
    if (idx < 49152) { int n = idx / 96, k = idx - n * 96;
      W2ih[idx] = (__bf16)(k < NIN ? l2Wih[n * NIN + k] : 0.f); continue; }
    idx -= 49152;
    if (idx < 65536) { W1hh[idx] = (__bf16)l1Whh[idx]; continue; }
    idx -= 65536;
    if (idx < 65536) { W2hh[idx] = (__bf16)l2Whh[idx]; continue; }
    idx -= 65536;
    if (idx < 65536) { DWhh[idx] = (__bf16)dWhh[idx]; continue; }
    idx -= 65536;
    if (idx < 16384) { int n = idx >> 7, k = idx & 127;
      W1d[idx] = (__bf16)decW1[n * 384 + k]; continue; }
    idx -= 16384;
    if (idx < 16384) { int n = idx >> 7, k = idx & 127;
      W1c[idx] = (__bf16)decW1[n * 384 + 128 + k]; continue; }
    idx -= 16384;
    { int n = idx >> 7, k = idx & 127;
      W1e[idx] = (__bf16)decW1[n * 384 + 256 + k]; }
  }
}

// ---------------------------------------------------------------------------
// Kernel 2: xc1 = XT@Wx1, xc2 = XT@Wx2, yc2 = y_prev@Wy2   (hoisted from scan)
// ---------------------------------------------------------------------------
__global__ __launch_bounds__(128) void dstp_precompute(
    const float* __restrict__ X, const float* __restrict__ y_prev,
    const float* __restrict__ attn1W, const float* __restrict__ attn2W,
    float* __restrict__ xc1, float* __restrict__ xc2, float* __restrict__ yc2)
{
  int b = blockIdx.x;
  int j = threadIdx.x;
  if (j < NINP) {
    float a1 = 0.f, a2 = 0.f;
    if (j < NIN) {
      for (int t = 0; t < TSTEPS; ++t) {
        float x = X[((size_t)b * TSTEPS + t) * NIN + j];
        a1 += x * attn1W[2 * HE + t];
        a2 += x * attn2W[2 * HE + t];
      }
    }
    xc1[(size_t)b * NINP + j] = a1;
    xc2[(size_t)b * NINP + j] = a2;
  } else if (j == NINP) {
    float a = 0.f;
    for (int t = 0; t < TSTEPS; ++t) a += y_prev[(size_t)b * TSTEPS + t] * attn2W[2 * HE + TSTEPS + t];
    yc2[b] = a;
  }
}

// ---------------------------------------------------------------------------
// gates(16x512) = A96(16x96)@Wih^T + A128(16x128)@Whh^T + bias
// 8 waves, each wave owns 4 column tiles of 16; per tile: 14 asm-streamed
// b128 loads, one dependency-carrying s_wait_loadcnt, then 7 WMMAs split
// across TWO independent accumulator chains (breaks the D->C serial chain).
// ---------------------------------------------------------------------------
__device__ __forceinline__ void gates_gemm(
    const __bf16* A96, const __bf16* A128,
    const __bf16* Wih, const __bf16* Whh,
    const float* bias, float* gatesOut, int tid)
{
  int wv = tid >> 5, lane = tid & 31;
  int m16 = lane & 15, hsel = (lane >> 4) & 1;
  v16bf ax0 = load_a(A96 + 0,  NINP, lane);
  v16bf ax1 = load_a(A96 + 32, NINP, lane);
  v16bf ax2 = load_a(A96 + 64, NINP, lane);
  v16bf ah0 = load_a(A128 + 0,  HE, lane);
  v16bf ah1 = load_a(A128 + 32, HE, lane);
  v16bf ah2 = load_a(A128 + 64, HE, lane);
  v16bf ah3 = load_a(A128 + 96, HE, lane);
#pragma unroll
  for (int ct = 0; ct < 4; ++ct) {
    int n0 = (wv + ct * 8) * 16;
    const __bf16* pW = Wih + (size_t)(n0 + m16) * NINP + hsel * 16;
    const __bf16* pH = Whh + (size_t)(n0 + m16) * HE   + hsel * 16;
    bfrag f0, f1, f2, f3, f4, f5, f6;
    issue_b(pW + 0,  f0);
    issue_b(pW + 32, f1);
    issue_b(pW + 64, f2);
    issue_b(pH + 0,  f3);
    issue_b(pH + 32, f4);
    issue_b(pH + 64, f5);
    issue_b(pH + 96, f6);
    wait_frags7(f0, f1, f2, f3, f4, f5, f6);
    v8f acc0 = {0.f,0.f,0.f,0.f,0.f,0.f,0.f,0.f};
    v8f acc1 = {0.f,0.f,0.f,0.f,0.f,0.f,0.f,0.f};
    acc0 = wmma_bf16(ax0, as_v16bf(f0), acc0);
    acc1 = wmma_bf16(ah0, as_v16bf(f3), acc1);
    acc0 = wmma_bf16(ax1, as_v16bf(f1), acc0);
    acc1 = wmma_bf16(ah1, as_v16bf(f4), acc1);
    acc0 = wmma_bf16(ax2, as_v16bf(f2), acc0);
    acc1 = wmma_bf16(ah2, as_v16bf(f5), acc1);
    acc1 = wmma_bf16(ah3, as_v16bf(f6), acc1);
    float bc = bias[n0 + m16];
#pragma unroll
    for (int v = 0; v < 8; ++v)
      gatesOut[(v + hsel * 8) * 512 + n0 + m16] = acc0[v] + acc1[v] + bc;
  }
}

// Elementwise LSTM cell update; refreshes bf16 mirrors for next step's WMMA A.
__device__ __forceinline__ void lstm_ew(const float* gates, float* h, float* c,
                                        __bf16* hb, __bf16* cb, int tid)
{
  for (int i = tid; i < BROWS * HE; i += 256) {
    int r = i >> 7, k = i & 127;
    const float* g = gates + r * 512;
    float gi = g[k], gf = g[128 + k], gg = g[256 + k], go = g[384 + k];
    float cn = sigf(gf) * c[i] + sigf(gi) * tanhf(gg);
    float hn = sigf(go) * tanhf(cn);
    c[i] = cn; h[i] = hn;
    hb[i] = (__bf16)hn;
    if (cb) cb[i] = (__bf16)cn;
  }
}

// ---------------------------------------------------------------------------
// Kernel 3: encoder — 64 blocks x 256 threads, 16 batch rows per block.
// ---------------------------------------------------------------------------
__global__ __launch_bounds__(256) void dstp_encoder(
    const float* __restrict__ X,
    const float* __restrict__ attn1W, const float* __restrict__ attn1b,
    const float* __restrict__ attn2W, const float* __restrict__ attn2b,
    const float* __restrict__ l1b, const float* __restrict__ l2b,
    const __bf16* W1ih, const __bf16* W1hh,
    const __bf16* W2ih, const __bf16* W2hh,
    const float* __restrict__ xc1g, const float* __restrict__ xc2g,
    const float* __restrict__ yc2g,
    __bf16* __restrict__ Xenc)
{
  __shared__ float  xt  [BROWS * NINP];
  __shared__ float  xt1 [BROWS * NINP];
  __shared__ __bf16 xt1b[BROWS * NINP];
  __shared__ __bf16 xt2b[BROWS * NINP];
  __shared__ float  hS [BROWS * HE], sS [BROWS * HE];
  __shared__ float  hsS[BROWS * HE], ssS[BROWS * HE];
  __shared__ __bf16 hB [BROWS * HE], hsB[BROWS * HE];
  __shared__ float  gates[BROWS * 512];
  __shared__ float  red[BROWS][16];
  __shared__ float  rrow[BROWS];

  const int tid = threadIdx.x;
  const int b0  = blockIdx.x * BROWS;
  const int wv  = tid >> 5, lane = tid & 31;

  for (int i = tid; i < BROWS * HE; i += 256) {
    hS[i] = sS[i] = hsS[i] = ssS[i] = 0.f;
    hB[i] = (__bf16)0.f; hsB[i] = (__bf16)0.f;
  }
  for (int i = tid; i < BROWS * NINP; i += 256) {
    xt1b[i] = (__bf16)0.f; xt2b[i] = (__bf16)0.f;  // K-pad lanes stay zero
  }
  __syncthreads();

  const float b1 = attn1b[0], b2 = attn2b[0];

  for (int t = 0; t < TSTEPS; ++t) {
    // ---- stage x_t tile via CDNA5 async copy (global -> LDS, ASYNCcnt)
    for (int i = tid; i < BROWS * NIN; i += 256) {
      int r = i / NIN, j = i - r * NIN;
      const float* gp = X + ((size_t)(b0 + r) * TSTEPS + t) * NIN + j;
      unsigned lofs = (unsigned)(uintptr_t)(&xt[r * NINP + j]);  // LDS aperture: low 32 bits
      asm volatile("global_load_async_to_lds_b32 %0, %1, off"
                   :: "v"(lofs), "v"(gp) : "memory");
    }
    asm volatile("s_wait_asynccnt 0x0" ::: "memory");
    __syncthreads();

    // ---- rrow[r] = h.Wh1 + s.Ws1
    {
      int r = tid >> 4, seg = tid & 15;
      float acc = 0.f;
      const float* hr = hS + r * HE; const float* sr = sS + r * HE;
#pragma unroll
      for (int q = 0; q < 8; ++q) {
        int k = seg * 8 + q;
        acc += hr[k] * attn1W[k] + sr[k] * attn1W[HE + k];
      }
      red[r][seg] = acc;
    }
    __syncthreads();
    if (tid < BROWS) {
      float acc = 0.f;
#pragma unroll
      for (int q = 0; q < 16; ++q) acc += red[tid][q];
      rrow[tid] = acc;
    }
    __syncthreads();

    // ---- input attention softmax over 81 features; x_tilde = alpha * x_t
    for (int rr = 0; rr < 2; ++rr) {
      int r = wv * 2 + rr;
      float base = rrow[r] + b1;
      const float* xcr = xc1g + (size_t)(b0 + r) * NINP;
      float v[3]; float mx = -3.0e38f;
#pragma unroll
      for (int q = 0; q < 3; ++q) {
        int j = lane + q * 32;
        v[q] = (j < NIN) ? (base + xcr[j]) : -3.0e38f;
        mx = fmaxf(mx, v[q]);
      }
      for (int off = 16; off > 0; off >>= 1) mx = fmaxf(mx, __shfl_xor(mx, off, 32));
      float e[3]; float sum = 0.f;
#pragma unroll
      for (int q = 0; q < 3; ++q) {
        int j = lane + q * 32;
        e[q] = (j < NIN) ? __expf(v[q] - mx) : 0.f;
        sum += e[q];
      }
      for (int off = 16; off > 0; off >>= 1) sum += __shfl_xor(sum, off, 32);
      float inv = 1.f / sum;
#pragma unroll
      for (int q = 0; q < 3; ++q) {
        int j = lane + q * 32;
        if (j < NIN) {
          float xv = e[q] * inv * xt[r * NINP + j];
          xt1[r * NINP + j]  = xv;
          xt1b[r * NINP + j] = (__bf16)xv;
        }
      }
    }
    __syncthreads();

    // ---- LSTM1
    gates_gemm(xt1b, hB, W1ih, W1hh, l1b, gates, tid);
    __syncthreads();
    lstm_ew(gates, hS, sS, hB, (__bf16*)nullptr, tid);
    __syncthreads();

    // ---- rrow[r] = hs.Wh2 + ss.Ws2
    {
      int r = tid >> 4, seg = tid & 15;
      float acc = 0.f;
      const float* hr = hsS + r * HE; const float* sr = ssS + r * HE;
#pragma unroll
      for (int q = 0; q < 8; ++q) {
        int k = seg * 8 + q;
        acc += hr[k] * attn2W[k] + sr[k] * attn2W[HE + k];
      }
      red[r][seg] = acc;
    }
    __syncthreads();
    if (tid < BROWS) {
      float acc = 0.f;
#pragma unroll
      for (int q = 0; q < 16; ++q) acc += red[tid][q];
      rrow[tid] = acc;
    }
    __syncthreads();

    // ---- second-stage attention; x_tilde2 = alpha2 * x_tilde
    for (int rr = 0; rr < 2; ++rr) {
      int r = wv * 2 + rr;
      float base = rrow[r] + yc2g[b0 + r] + b2;
      const float* xcr = xc2g + (size_t)(b0 + r) * NINP;
      float v[3]; float mx = -3.0e38f;
#pragma unroll
      for (int q = 0; q < 3; ++q) {
        int j = lane + q * 32;
        v[q] = (j < NIN) ? (base + xcr[j]) : -3.0e38f;
        mx = fmaxf(mx, v[q]);
      }
      for (int off = 16; off > 0; off >>= 1) mx = fmaxf(mx, __shfl_xor(mx, off, 32));
      float e[3]; float sum = 0.f;
#pragma unroll
      for (int q = 0; q < 3; ++q) {
        int j = lane + q * 32;
        e[q] = (j < NIN) ? __expf(v[q] - mx) : 0.f;
        sum += e[q];
      }
      for (int off = 16; off > 0; off >>= 1) sum += __shfl_xor(sum, off, 32);
      float inv = 1.f / sum;
#pragma unroll
      for (int q = 0; q < 3; ++q) {
        int j = lane + q * 32;
        if (j < NIN) {
          float xv = e[q] * inv * xt1[r * NINP + j];
          xt2b[r * NINP + j] = (__bf16)xv;
        }
      }
    }
    __syncthreads();

    // ---- LSTM2
    gates_gemm(xt2b, hsB, W2ih, W2hh, l2b, gates, tid);
    __syncthreads();
    lstm_ew(gates, hsS, ssS, hsB, (__bf16*)nullptr, tid);
    __syncthreads();

    // ---- emit X_encoded[:, t, :]
    for (int i = tid; i < BROWS * HE; i += 256) {
      int r = i >> 7, e2 = i & 127;
      Xenc[((size_t)(b0 + r) * TSTEPS + t) * HE + e2] = (__bf16)hsS[i];
    }
    __syncthreads();
  }
}

// ---------------------------------------------------------------------------
// Kernel 4: Epre = X_encoded @ W1e^T + dec_b1   (32768x128x128 GEMM, bf16 out)
// ---------------------------------------------------------------------------
__global__ __launch_bounds__(256) void dstp_epre(
    const __bf16* __restrict__ Xenc, const __bf16* __restrict__ W1e,
    const float* __restrict__ decb1, __bf16* __restrict__ Epre)
{
  const int row0 = blockIdx.x * 16;     // over B*TSTEPS rows
  const int wv = threadIdx.x >> 5, lane = threadIdx.x & 31;
  const int m16 = lane & 15, hsel = (lane >> 4) & 1;
  const int n0 = wv * 16;
  v8f acc0 = {0.f,0.f,0.f,0.f,0.f,0.f,0.f,0.f};
  v8f acc1 = {0.f,0.f,0.f,0.f,0.f,0.f,0.f,0.f};
#pragma unroll
  for (int kc = 0; kc < 4; kc += 2) {
    v16bf a0 = load_a(Xenc + (size_t)row0 * HE + kc * 32, HE, lane);
    v16bf b0 = load_b(W1e + (size_t)n0 * HE + kc * 32, HE, lane);
    v16bf a1 = load_a(Xenc + (size_t)row0 * HE + (kc + 1) * 32, HE, lane);
    v16bf b1 = load_b(W1e + (size_t)n0 * HE + (kc + 1) * 32, HE, lane);
    acc0 = wmma_bf16(a0, b0, acc0);
    acc1 = wmma_bf16(a1, b1, acc1);
  }
#pragma unroll
  for (int v = 0; v < 8; ++v) {
    int row = row0 + v + hsel * 8;
    int n = n0 + m16;
    Epre[(size_t)row * HE + n] = (__bf16)(acc0[v] + acc1[v] + decb1[n]);
  }
}

// ---------------------------------------------------------------------------
// Kernel 5: decoder — 64 blocks x 256 threads, 16 batch rows per block.
// ---------------------------------------------------------------------------
__global__ __launch_bounds__(256) void dstp_decoder(
    const float* __restrict__ y_prev,
    const float* __restrict__ decW2, const float* __restrict__ decb2,
    const float* __restrict__ dlWih, const float* __restrict__ dlb,
    const float* __restrict__ fcW,   const float* __restrict__ fcb,
    const float* __restrict__ fcfW,  const float* __restrict__ fcfb,
    const __bf16* W1d,  const __bf16* W1c, const __bf16* DWhh,
    const __bf16* Xenc, const __bf16* Epre,
    float* __restrict__ out)
{
  __shared__ float  dS[BROWS * HE], cS[BROWS * HE];
  __shared__ __bf16 dB[BROWS * HE], cB[BROWS * HE];
  __shared__ float  uS[BROWS * HE];
  __shared__ float  gates[BROWS * 512];
  __shared__ float  ctx[BROWS * HE];
  __shared__ float  score[BROWS][TSTEPS];
  __shared__ float  red[BROWS][16];
  __shared__ float  ytS[BROWS];
  __shared__ float  w2L[HE];

  const int tid = threadIdx.x;
  const int b0  = blockIdx.x * BROWS;
  const int wv  = tid >> 5, lane = tid & 31;
  const int m16 = lane & 15, hsel = (lane >> 4) & 1;

  for (int i = tid; i < BROWS * HE; i += 256) {
    dS[i] = cS[i] = 0.f; dB[i] = (__bf16)0.f; cB[i] = (__bf16)0.f;
  }
  if (tid < HE) w2L[tid] = decW2[tid];
  __syncthreads();

  const float db2  = decb2[0];
  const float fcy  = fcW[HE];
  const float fcb0 = fcb[0];

  for (int t = 0; t < TSTEPS; ++t) {
    // ---- u = d@W1d^T + c@W1c^T  (one 16-col tile per wave, 8 WMMAs, 2 chains)
    {
      v16bf ad0 = load_a(dB + 0,  HE, lane), ad1 = load_a(dB + 32, HE, lane);
      v16bf ad2 = load_a(dB + 64, HE, lane), ad3 = load_a(dB + 96, HE, lane);
      v16bf ac0 = load_a(cB + 0,  HE, lane), ac1 = load_a(cB + 32, HE, lane);
      v16bf ac2 = load_a(cB + 64, HE, lane), ac3 = load_a(cB + 96, HE, lane);
      int n0 = wv * 16;
      const __bf16* pD = W1d + (size_t)(n0 + m16) * HE + hsel * 16;
      const __bf16* pC = W1c + (size_t)(n0 + m16) * HE + hsel * 16;
      bfrag f0, f1, f2, f3, f4, f5, f6, f7;
      issue_b(pD + 0,  f0); issue_b(pD + 32, f1);
      issue_b(pD + 64, f2); issue_b(pD + 96, f3);
      issue_b(pC + 0,  f4); issue_b(pC + 32, f5);
      issue_b(pC + 64, f6); issue_b(pC + 96, f7);
      wait_frags8(f0, f1, f2, f3, f4, f5, f6, f7);
      v8f acc0 = {0.f,0.f,0.f,0.f,0.f,0.f,0.f,0.f};
      v8f acc1 = {0.f,0.f,0.f,0.f,0.f,0.f,0.f,0.f};
      acc0 = wmma_bf16(ad0, as_v16bf(f0), acc0);
      acc1 = wmma_bf16(ac0, as_v16bf(f4), acc1);
      acc0 = wmma_bf16(ad1, as_v16bf(f1), acc0);
      acc1 = wmma_bf16(ac1, as_v16bf(f5), acc1);
      acc0 = wmma_bf16(ad2, as_v16bf(f2), acc0);
      acc1 = wmma_bf16(ac2, as_v16bf(f6), acc1);
      acc0 = wmma_bf16(ad3, as_v16bf(f3), acc0);
      acc1 = wmma_bf16(ac3, as_v16bf(f7), acc1);
#pragma unroll
      for (int v = 0; v < 8; ++v)
        uS[(v + hsel * 8) * HE + n0 + m16] = acc0[v] + acc1[v];
    }
    __syncthreads();

    // ---- score[r][tt] = tanh(Epre + u) . dec_W2 + dec_b2
    for (int pi = tid; pi < BROWS * TSTEPS; pi += 256) {
      int r = pi >> 5, tt = pi & 31;
      const __bf16* ep = Epre + ((size_t)(b0 + r) * TSTEPS + tt) * HE;
      const float* ur = uS + r * HE;
      float acc = 0.f;
      for (int h = 0; h < HE; ++h)
        acc += tanhf((float)ep[h] + ur[h]) * w2L[h];
      score[r][tt] = acc + db2;
    }
    __syncthreads();

    // ---- softmax over time (32 lanes = 32 timesteps)
    for (int rr = 0; rr < 2; ++rr) {
      int r = wv * 2 + rr;
      float v = score[r][lane];
      float mx = v;
      for (int off = 16; off > 0; off >>= 1) mx = fmaxf(mx, __shfl_xor(mx, off, 32));
      float e = __expf(v - mx);
      float sum = e;
      for (int off = 16; off > 0; off >>= 1) sum += __shfl_xor(sum, off, 32);
      score[r][lane] = e / sum;
    }
    __syncthreads();

    // ---- context[r] = sum_t beta[r][t] * Xenc[b][t][:]
    {
      int r = tid >> 4, e0 = (tid & 15) * 8;
      float acc8[8] = {0.f,0.f,0.f,0.f,0.f,0.f,0.f,0.f};
      const __bf16* xr = Xenc + (size_t)(b0 + r) * TSTEPS * HE;
      for (int tt = 0; tt < TSTEPS; ++tt) {
        float bta = score[r][tt];
        const __bf16* xp = xr + tt * HE + e0;
#pragma unroll
        for (int q = 0; q < 8; ++q) acc8[q] += bta * (float)xp[q];
      }
#pragma unroll
      for (int q = 0; q < 8; ++q) ctx[r * HE + e0 + q] = acc8[q];
    }
    __syncthreads();

    // ---- y_tilde[r] = ctx . fc_c + y_prev[b][t] * fc_y + fc_b
    {
      int r = tid >> 4, seg = tid & 15;
      float acc = 0.f;
#pragma unroll
      for (int q = 0; q < 8; ++q) {
        int k = seg * 8 + q;
        acc += ctx[r * HE + k] * fcW[k];
      }
      red[r][seg] = acc;
    }
    __syncthreads();
    if (tid < BROWS) {
      float acc = 0.f;
#pragma unroll
      for (int q = 0; q < 16; ++q) acc += red[tid][q];
      ytS[tid] = acc + y_prev[(size_t)(b0 + tid) * TSTEPS + t] * fcy + fcb0;
    }
    __syncthreads();

    // ---- dlstm gates: rank-1 (y_tilde x dlWih) + d@DWhh^T + bias
    {
      v16bf ad0 = load_a(dB + 0,  HE, lane), ad1 = load_a(dB + 32, HE, lane);
      v16bf ad2 = load_a(dB + 64, HE, lane), ad3 = load_a(dB + 96, HE, lane);
#pragma unroll
      for (int ct = 0; ct < 4; ++ct) {
        int n0 = (wv + ct * 8) * 16;
        const __bf16* pH = DWhh + (size_t)(n0 + m16) * HE + hsel * 16;
        bfrag f0, f1, f2, f3;
        issue_b(pH + 0,  f0); issue_b(pH + 32, f1);
        issue_b(pH + 64, f2); issue_b(pH + 96, f3);
        wait_frags4(f0, f1, f2, f3);
        v8f acc0 = {0.f,0.f,0.f,0.f,0.f,0.f,0.f,0.f};
        v8f acc1 = {0.f,0.f,0.f,0.f,0.f,0.f,0.f,0.f};
        acc0 = wmma_bf16(ad0, as_v16bf(f0), acc0);
        acc1 = wmma_bf16(ad1, as_v16bf(f1), acc1);
        acc0 = wmma_bf16(ad2, as_v16bf(f2), acc0);
        acc1 = wmma_bf16(ad3, as_v16bf(f3), acc1);
        float wn = dlWih[n0 + m16];
        float bn = dlb[n0 + m16];
#pragma unroll
        for (int v = 0; v < 8; ++v) {
          int row = v + hsel * 8;
          gates[row * 512 + n0 + m16] = acc0[v] + acc1[v] + bn + ytS[row] * wn;
        }
      }
    }
    __syncthreads();
    lstm_ew(gates, dS, cS, dB, cB, tid);
    __syncthreads();
  }

  // ---- y_pred = [d_fin, context] @ fcf_W^T + fcf_b
  {
    int r = tid >> 4, seg = tid & 15;
    float acc = 0.f;
#pragma unroll
    for (int q = 0; q < 16; ++q) {
      int k = seg * 16 + q;
      acc += (k < HE) ? dS[r * HE + k] * fcfW[k]
                      : ctx[r * HE + (k - HE)] * fcfW[k];
    }
    red[r][seg] = acc;
  }
  __syncthreads();
  if (tid < BROWS) {
    float acc = 0.f;
#pragma unroll
    for (int q = 0; q < 16; ++q) acc += red[tid][q];
    out[b0 + tid] = acc + fcfb[0];
  }
}

// ---------------------------------------------------------------------------
// Host-side launcher
// ---------------------------------------------------------------------------
extern "C" void kernel_launch(void* const* d_in, const int* in_sizes, int n_in,
                              void* d_out, int out_size, void* d_ws, size_t ws_size,
                              hipStream_t stream) {
  const float* X      = (const float*)d_in[0];
  const float* y_prev = (const float*)d_in[1];
  const float* attn1W = (const float*)d_in[2];
  const float* attn1b = (const float*)d_in[3];
  const float* attn2W = (const float*)d_in[4];
  const float* attn2b = (const float*)d_in[5];
  const float* l1Wih  = (const float*)d_in[6];
  const float* l1Whh  = (const float*)d_in[7];
  const float* l1b    = (const float*)d_in[8];
  const float* l2Wih  = (const float*)d_in[9];
  const float* l2Whh  = (const float*)d_in[10];
  const float* l2b    = (const float*)d_in[11];
  const float* decW1  = (const float*)d_in[12];
  const float* decb1  = (const float*)d_in[13];
  const float* decW2  = (const float*)d_in[14];
  const float* decb2  = (const float*)d_in[15];
  const float* dlWih  = (const float*)d_in[16];
  const float* dlWhh  = (const float*)d_in[17];
  const float* dlb    = (const float*)d_in[18];
  const float* fcW    = (const float*)d_in[19];
  const float* fcb    = (const float*)d_in[20];
  const float* fcfW   = (const float*)d_in[21];
  const float* fcfb   = (const float*)d_in[22];
  float* out = (float*)d_out;

  char* ws = (char*)d_ws;
  __bf16* W1ih = (__bf16*)(ws + 0);          // 512x96
  __bf16* W1hh = (__bf16*)(ws + 98304);      // 512x128
  __bf16* W2ih = (__bf16*)(ws + 229376);
  __bf16* W2hh = (__bf16*)(ws + 327680);
  __bf16* DWhh = (__bf16*)(ws + 458752);
  __bf16* W1e  = (__bf16*)(ws + 589824);     // 128x128
  __bf16* W1d  = (__bf16*)(ws + 622592);
  __bf16* W1c  = (__bf16*)(ws + 655360);
  float*  xc1  = (float*)(ws + 688128);      // B x 96
  float*  xc2  = (float*)(ws + 1081344);
  float*  yc2  = (float*)(ws + 1474560);     // B
  __bf16* Xenc = (__bf16*)(ws + 1478656);    // B x 32 x 128
  __bf16* Epre = (__bf16*)(ws + 9867264);    // B*32 x 128

  dstp_prep<<<1344, 256, 0, stream>>>(l1Wih, l1Whh, l2Wih, l2Whh, dlWhh, decW1,
                                      W1ih, W1hh, W2ih, W2hh, DWhh, W1e, W1d, W1c);
  dstp_precompute<<<BSZ, 128, 0, stream>>>(X, y_prev, attn1W, attn2W, xc1, xc2, yc2);
  dstp_encoder<<<BSZ / BROWS, 256, 0, stream>>>(X, attn1W, attn1b, attn2W, attn2b,
                                                l1b, l2b, W1ih, W1hh, W2ih, W2hh,
                                                xc1, xc2, yc2, Xenc);
  dstp_epre<<<(BSZ * TSTEPS) / 16, 256, 0, stream>>>(Xenc, W1e, decb1, Epre);
  dstp_decoder<<<BSZ / BROWS, 256, 0, stream>>>(y_prev, decW2, decb2, dlWih, dlb,
                                                fcW, fcb, fcfW, fcfb,
                                                W1d, W1c, DWhh, Xenc, Epre, out);
}
